// Code2VecEncoder_51470888075930
// MI455X (gfx1250) — matrix operand order repaired and checked by reading, not compiled, not run on hardware
//
#include <hip/hip_runtime.h>
#include <hip/hip_bf16.h>
#include <math.h>

// ---- problem constants (from reference) ----
#define Bn   512
#define Cn   200
#define En   128
#define CVn  384
#define K3n  384              // 3*E
#define ROWS (Bn*Cn)          // 102400
#define ASTRIDE 392           // padded LDS row stride (halfwords): 392%64(dw)=4 -> bank spread

typedef __bf16 v16bf __attribute__((ext_vector_type(16)));
typedef float  v8f   __attribute__((ext_vector_type(8)));

union FragBF { v16bf v; uint4 q[2]; };

__device__ __forceinline__ unsigned short f2bf(float f) {
  unsigned int u = __float_as_uint(f);
  u += 0x7FFFu + ((u >> 16) & 1u);        // round-to-nearest-even
  return (unsigned short)(u >> 16);
}
__device__ __forceinline__ float bf2f(unsigned short h) {
  return __uint_as_float(((unsigned int)h) << 16);
}

// ---------------- kernel 0: W f32 -> bf16 (row-major [v][k]) ----------------
__global__ void k0_convW(const float* __restrict__ W, unsigned short* __restrict__ Wb) {
  int t = blockIdx.x * blockDim.x + threadIdx.x;
  if (t < CVn * K3n) Wb[t] = f2bf(W[t]);
}

// ------------- kernel 1: gather + bf16 WMMA GEMM + tanh -> h(bf16) -------------
// grid = ROWS/16 = 6400 blocks, 256 threads (8 waves). Each block: 16 rows x 384 cols.
__global__ __launch_bounds__(256) void k1_gemm(
    const int* __restrict__ starts, const int* __restrict__ paths,
    const int* __restrict__ ends,
    const float* __restrict__ node_emb, const float* __restrict__ path_emb,
    const unsigned short* __restrict__ Wb, unsigned short* __restrict__ hbuf)
{
  __shared__ unsigned short Atile[16 * ASTRIDE];
  __shared__ int sIdx[16], pIdx[16], eIdx[16];
  const int tid = threadIdx.x;
  const int rowBase = blockIdx.x * 16;

  if (tid < 16) {
    sIdx[tid] = starts[rowBase + tid];
    pIdx[tid] = paths [rowBase + tid];
    eIdx[tid] = ends  [rowBase + tid];
  }
  __syncthreads();

  // Gather ctx tile [16 x 384] f32 -> bf16 LDS (coalesced 128-element runs)
  for (int idx = tid; idx < 16 * K3n; idx += 256) {
    int row = idx / K3n;
    int k   = idx - row * K3n;
    int kk  = k & 127;
    float v;
    if      (k < 128) v = node_emb[(size_t)sIdx[row] * En + kk];
    else if (k < 256) v = path_emb[(size_t)pIdx[row] * En + kk];
    else              v = node_emb[(size_t)eIdx[row] * En + kk];
    Atile[row * ASTRIDE + k] = f2bf(v);
  }
  __syncthreads();

  const int lane   = tid & 31;
  const int wave   = tid >> 5;
  const int laneHi = lane >> 4;
  const int m      = lane & 15;   // A-row for src, B/D-column index

  v8f z = {0.f,0.f,0.f,0.f,0.f,0.f,0.f,0.f};
  v8f acc0 = z, acc1 = z, acc2 = z;
  const int ct0 = wave, ct1 = wave + 8, ct2 = wave + 16;   // 3 N-tiles per wave

  for (int kb = 0; kb < 12; ++kb) {
    const int k0 = kb * 32;

    // A fragment: 16x32 bf16, contiguous k order, same as B
    //  lanes 0-15 : K = k0..k0+15 ; lanes 16-31: K = k0+16..k0+31
    FragBF af;
    const uint4* ap = (const uint4*)&Atile[m * ASTRIDE + k0 + laneHi * 16];
    af.q[0] = ap[0];        // 8 halfwords
    af.q[1] = ap[1];        // +8 halfwords

    // B fragment: lane n holds column n, 16 consecutive K (halves split at 16)
    FragBF bf;
    const uint4* bp0 = (const uint4*)&Wb[(size_t)(ct0*16 + m) * K3n + k0 + laneHi*16];
    bf.q[0] = bp0[0]; bf.q[1] = bp0[1];
    acc0 = __builtin_amdgcn_wmma_f32_16x16x32_bf16(false, af.v, false, bf.v,
                                                   (short)0, acc0, false, false);

    const uint4* bp1 = (const uint4*)&Wb[(size_t)(ct1*16 + m) * K3n + k0 + laneHi*16];
    bf.q[0] = bp1[0]; bf.q[1] = bp1[1];
    acc1 = __builtin_amdgcn_wmma_f32_16x16x32_bf16(false, af.v, false, bf.v,
                                                   (short)0, acc1, false, false);

    const uint4* bp2 = (const uint4*)&Wb[(size_t)(ct2*16 + m) * K3n + k0 + laneHi*16];
    bf.q[0] = bp2[0]; bf.q[1] = bp2[1];
    acc2 = __builtin_amdgcn_wmma_f32_16x16x32_bf16(false, af.v, false, bf.v,
                                                   (short)0, acc2, false, false);
  }

  // Epilogue: tanh, store h as bf16. D layout: VGPR j -> row j (lanes 0-15) / j+8 (16-31)
  const int col0 = ct0*16 + m, col1 = ct1*16 + m, col2 = ct2*16 + m;
  #pragma unroll
  for (int j = 0; j < 8; ++j) {
    int grow = rowBase + j + laneHi * 8;
    size_t base = (size_t)grow * CVn;
    hbuf[base + col0] = f2bf(tanhf(acc0[j]));
    hbuf[base + col1] = f2bf(tanhf(acc1[j]));
    hbuf[base + col2] = f2bf(tanhf(acc2[j]));
  }
}

// ---------------- kernel 2: scores[r] = h[r,:] . a ----------------
// grid = ROWS/8, 256 threads: one wave per row.
__global__ __launch_bounds__(256) void k2_scores(
    const unsigned short* __restrict__ hbuf, const float* __restrict__ a,
    float* __restrict__ scores)
{
  const int lane = threadIdx.x & 31;
  const int wave = threadIdx.x >> 5;
  const int row  = blockIdx.x * 8 + wave;
  const unsigned short* hr = hbuf + (size_t)row * CVn;
  float acc = 0.f;
  #pragma unroll
  for (int v = lane; v < CVn; v += 32) acc += bf2f(hr[v]) * a[v];
  for (int off = 16; off; off >>= 1) acc += __shfl_xor(acc, off, 32);
  if (lane == 0) scores[row] = acc;
}

// ---------------- kernel 3: softmax over contexts (axis c) ----------------
// grid = B, 256 threads.
__global__ __launch_bounds__(256) void k3_softmax(
    const float* __restrict__ scores, float* __restrict__ attn)
{
  const int b = blockIdx.x, tid = threadIdx.x;
  const int lane = tid & 31, wave = tid >> 5;
  __shared__ float red[8];

  float s0 = (tid < Cn) ? scores[b * Cn + tid] : -3.4e38f;
  float m = s0;
  for (int off = 16; off; off >>= 1) m = fmaxf(m, __shfl_xor(m, off, 32));
  if (lane == 0) red[wave] = m;
  __syncthreads();
  m = red[0];
  #pragma unroll
  for (int w = 1; w < 8; ++w) m = fmaxf(m, red[w]);

  float e = (tid < Cn) ? expf(s0 - m) : 0.f;
  float sum = e;
  for (int off = 16; off; off >>= 1) sum += __shfl_xor(sum, off, 32);
  __syncthreads();
  if (lane == 0) red[wave] = sum;
  __syncthreads();
  sum = 0.f;
  #pragma unroll
  for (int w = 0; w < 8; ++w) sum += red[w];

  if (tid < Cn) attn[b * Cn + tid] = e / sum;
}

// ---------------- kernel 4: out[b,v] = sum_c h[b,c,v] * attn[b,c] ----------------
// grid = B, 384 threads (one per v), coalesced over v.
__global__ __launch_bounds__(384) void k4_wsum(
    const unsigned short* __restrict__ hbuf, const float* __restrict__ attn,
    float* __restrict__ out)
{
  const int b = blockIdx.x, v = threadIdx.x;
  __shared__ float at[Cn];
  if (v < Cn) at[v] = attn[b * Cn + v];
  __syncthreads();

  float acc = 0.f;
  const unsigned short* hb = hbuf + (size_t)b * Cn * CVn + v;
  #pragma unroll 4
  for (int c = 0; c < Cn; ++c) acc += bf2f(hb[(size_t)c * CVn]) * at[c];
  out[b * CVn + v] = acc;
}

// ---------------- launcher ----------------
extern "C" void kernel_launch(void* const* d_in, const int* in_sizes, int n_in,
                              void* d_out, int out_size, void* d_ws, size_t ws_size,
                              hipStream_t stream) {
  const int*   starts   = (const int*)  d_in[0];
  const int*   paths    = (const int*)  d_in[1];
  const int*   ends     = (const int*)  d_in[2];
  // d_in[3] = masks (unused by reference)
  const float* node_emb = (const float*)d_in[4];
  const float* path_emb = (const float*)d_in[5];
  const float* W        = (const float*)d_in[6];
  const float* a        = (const float*)d_in[7];
  float*       out      = (float*)d_out;

  char* ws = (char*)d_ws;
  const size_t WB_BYTES = (size_t)CVn * K3n * 2;            //   294,912
  const size_t H_BYTES  = (size_t)ROWS * CVn * 2;           // 78,643,200
  unsigned short* Wb     = (unsigned short*)ws;
  unsigned short* hbuf   = (unsigned short*)(ws + WB_BYTES);
  float*          scores = (float*)(ws + WB_BYTES + H_BYTES);
  float*          attn   = scores + ROWS;

  k0_convW  <<<(CVn * K3n + 255) / 256, 256, 0, stream>>>(W, Wb);
  k1_gemm   <<<ROWS / 16, 256, 0, stream>>>(starts, paths, ends,
                                            node_emb, path_emb, Wb, hbuf);
  k2_scores <<<ROWS / 8, 256, 0, stream>>>(hbuf, a, scores);
  k3_softmax<<<Bn, 256, 0, stream>>>(scores, attn);
  k4_wsum   <<<Bn, 384, 0, stream>>>(hbuf, attn, out);
}
